// LightGATIP_80599356277299
// MI455X (gfx1250) — matrix operand order repaired and checked by reading, not compiled, not run on hardware
//
#include <hip/hip_runtime.h>
#include <math.h>

typedef __attribute__((ext_vector_type(16))) _Float16 v16h;
typedef __attribute__((ext_vector_type(8)))  float    v8f;

#define DDIM 64
#define NEG_SLOPE 0.2f
#define DECAY 1e-4f

__device__ __forceinline__ void atomicMaxF(float* addr, float v) {
    // float atomic-max via sign-split int/uint atomics; addr initialized to -inf.
    if (v >= 0.0f) atomicMax((int*)addr, __float_as_int(v));
    else           atomicMin((unsigned int*)addr, __float_as_uint(v));
}

// 16 batched 64-dim f32 dot products per wave32 via two v_wmma_f32_16x16x32_f16.
// arow/brow: this lane's row pointers for row/col index (lane & 15).
// Returns dot(A_m, B_m) on lanes m<16 (duplicated on lanes 16..31).
// ldsW: 256 floats of per-wave scratch. Must be called uniformly by the whole block.
__device__ __forceinline__ float wave_dot16(const float* __restrict__ arow,
                                            const float* __restrict__ brow,
                                            float* __restrict__ ldsW) {
    const int lane = threadIdx.x & 31;
    const int half = lane >> 4;
    v8f c = {};
    #pragma unroll
    for (int chunk = 0; chunk < 2; ++chunk) {
        const int base = chunk * 32;
        v16h a, b;
        // 16-bit A 16x32 layout: lanes<16 hold K=0..7 (v0-3) & 16..23 (v4-7),
        // lanes>=16 hold K=8..15 & 24..31, row M = lane&15.
        #pragma unroll
        for (int j = 0; j < 8; ++j) {
            a[j]     = (_Float16)arow[base + half * 8 + j];
            a[8 + j] = (_Float16)arow[base + 16 + half * 8 + j];
        }
        // 16-bit B 32x16 layout: lane = column N (&15), lanes<16 hold K=0..7 & 16..23,
        // lanes>=16 hold K=8..15 & 24..31 (same K order as A).
        #pragma unroll
        for (int j = 0; j < 8; ++j) {
            b[j]     = (_Float16)brow[base + half * 8 + j];
            b[8 + j] = (_Float16)brow[base + 16 + half * 8 + j];
        }
        c = __builtin_amdgcn_wmma_f32_16x16x32_f16(false, a, false, b,
                                                   (short)0, c, false, false);
    }
    // Spill 16x16 result tile to LDS, read back the diagonal.
    #pragma unroll
    for (int r = 0; r < 8; ++r) ldsW[lane * 8 + r] = c[r];
    __syncthreads();
    const int m    = lane & 15;
    const int lsrc = (m < 8) ? m : (m + 16);  // D layout: (M=m,N=m) source lane
    float v = ldsW[lsrc * 8 + (m & 7)];
    __syncthreads();
    return v;
}

// ---------------- init / reset ----------------

__global__ void k_init(const float* __restrict__ uemb, const float* __restrict__ iemb,
                       float* ucur, float* usum, float* icur, float* isum,
                       float* acc, int nu64, int ni64) {
    int t = blockIdx.x * blockDim.x + threadIdx.x;
    if (t < nu64) { float v = uemb[t]; ucur[t] = v; usum[t] = v; }
    if (t < ni64) { float v = iemb[t]; icur[t] = v; isum[t] = v; }
    if (t < 2) acc[t] = 0.0f;
}

__global__ void k_reset(float* unxt, float* inxt,
                        float* umax, float* uden, float* imax, float* iden,
                        int nu, int ni, int nu64, int ni64) {
    int t = blockIdx.x * blockDim.x + threadIdx.x;
    if (t < nu64) unxt[t] = 0.0f;
    if (t < ni64) inxt[t] = 0.0f;
    if (t < nu) { umax[t] = -INFINITY; uden[t] = 0.0f; }
    if (t < ni) { imax[t] = -INFINITY; iden[t] = 0.0f; }
}

// ---------------- pass 1: edge scores via WMMA + segment max ----------------

__global__ void k_pass1(const float* __restrict__ ucur, const float* __restrict__ icur,
                        const int* __restrict__ uidx, const int* __restrict__ iidx,
                        float* __restrict__ a_e, float* umax, float* imax, int E) {
    __shared__ float ldsC[8 * 256];
    const int wave = threadIdx.x >> 5, lane = threadIdx.x & 31;
    const long k0 = ((long)blockIdx.x * 8 + wave) * 16;
    const int r = lane & 15;
    long kr = k0 + r; if (kr > E - 1) kr = E - 1;   // clamp: keep EXEC all-ones for WMMA
    const int em = (int)kr;
    const int u = uidx[em], i = iidx[em];
    const float* arow = ucur + (size_t)u * DDIM;
    const float* brow = icur + (size_t)i * DDIM;
    float s = wave_dot16(arow, brow, ldsC + wave * 256);
    if (lane < 16) {
        long km = k0 + lane; if (km > E - 1) km = E - 1;
        const int e2 = (int)km;
        float a = s > 0.0f ? s : NEG_SLOPE * s;     // leaky_relu
        a_e[e2] = a;
        atomicMaxF(umax + uidx[e2], a);
        atomicMaxF(imax + iidx[e2], a);
    }
}

// ---------------- pass 2: exp + denominators ----------------

__global__ void k_pass2(const float* __restrict__ a_e,
                        const int* __restrict__ uidx, const int* __restrict__ iidx,
                        const float* __restrict__ umax, const float* __restrict__ imax,
                        float* uden, float* iden,
                        float* __restrict__ exu, float* __restrict__ exi, int E) {
    int k = blockIdx.x * blockDim.x + threadIdx.x;
    if (k >= E) return;
    float a = a_e[k];
    int u = uidx[k], i = iidx[k];
    float eu = expf(a - umax[u]);
    float ei = expf(a - imax[i]);
    exu[k] = eu; exi[k] = ei;
    atomicAdd(uden + u, eu);
    atomicAdd(iden + i, ei);
}

// ---------------- pass 3: weighted scatter (edge,dim) ----------------

__global__ void k_pass3(const float* __restrict__ exu, const float* __restrict__ exi,
                        const float* __restrict__ uden, const float* __restrict__ iden,
                        const int* __restrict__ uidx, const int* __restrict__ iidx,
                        const float* __restrict__ ucur, const float* __restrict__ icur,
                        float* unxt, float* inxt, int E) {
    int gid = blockIdx.x * blockDim.x + threadIdx.x;
    int k = gid >> 6, d = gid & 63;
    if (k >= E) return;
    int u = uidx[k], i = iidx[k];
    float wu = exu[k] / uden[u];
    float wi = exi[k] / iden[i];
    atomicAdd(&unxt[(size_t)u * DDIM + d], wu * icur[(size_t)i * DDIM + d]);
    atomicAdd(&inxt[(size_t)i * DDIM + d], wi * ucur[(size_t)u * DDIM + d]);
}

// ---------------- accumulate running mean-sum ----------------

__global__ void k_accum(float* usum, float* isum,
                        const float* __restrict__ unxt, const float* __restrict__ inxt,
                        int nu64, int ni64) {
    int t = blockIdx.x * blockDim.x + threadIdx.x;
    if (t < nu64) usum[t] += unxt[t];
    if (t < ni64) isum[t] += inxt[t];
}

// ---------------- loss: WMMA pos/neg scores + reduction ----------------

__global__ void k_loss(const float* __restrict__ usum, const float* __restrict__ isum,
                       const int* __restrict__ pos, const int* __restrict__ neg,
                       float* acc, int NU) {
    __shared__ float ldsC[8 * 256];
    const int wave = threadIdx.x >> 5, lane = threadIdx.x & 31;
    const int ubase = (blockIdx.x * 8 + wave) * 16;
    const int r = lane & 15;
    int uid = ubase + r; if (uid > NU - 1) uid = NU - 1;   // clamp for WMMA EXEC
    const float* ur = usum + (size_t)uid * DDIM;
    const float* pr = isum + (size_t)pos[uid] * DDIM;
    const float* nr = isum + (size_t)neg[uid] * DDIM;
    float sp = wave_dot16(ur, pr, ldsC + wave * 256);
    float sn = wave_dot16(ur, nr, ldsC + wave * 256);
    if (lane < 16 && (ubase + lane) < NU) {
        float un = 0.f, pn = 0.f, nn = 0.f;
        #pragma unroll 8
        for (int d = 0; d < DDIM; ++d) {
            un += ur[d] * ur[d]; pn += pr[d] * pr[d]; nn += nr[d] * nr[d];
        }
        // embeddings are sums of 3 layers; mean => /3, dots => /9
        const float inv9 = 1.0f / 9.0f;
        float x  = (sn - sp) * inv9;
        float mf = fmaxf(x, 0.0f) + log1pf(expf(-fabsf(x)));   // stable softplus
        float rg = 0.5f * (un + pn + nn) * inv9;
        atomicAdd(acc + 0, mf);
        atomicAdd(acc + 1, rg);
    }
}

__global__ void k_final(const float* __restrict__ acc, float* out, int NU) {
    if (threadIdx.x == 0 && blockIdx.x == 0) {
        out[0] = acc[0] / (float)NU;             // mf_loss (mean over users)
        out[1] = DECAY * acc[1] / (float)NU;     // emb_loss (BATCH_SIZE == NU)
    }
}

// ---------------- host orchestration ----------------

static inline size_t alignup(size_t x) { return (x + 255) & ~(size_t)255; }

extern "C" void kernel_launch(void* const* d_in, const int* in_sizes, int n_in,
                              void* d_out, int out_size, void* d_ws, size_t ws_size,
                              hipStream_t stream) {
    const float* uemb = (const float*)d_in[0];
    const float* iemb = (const float*)d_in[1];
    const int*   uidx = (const int*)d_in[2];
    const int*   iidx = (const int*)d_in[3];
    const int*   posi = (const int*)d_in[4];
    const int*   negi = (const int*)d_in[5];
    float* out = (float*)d_out;

    const int NU = in_sizes[0] / DDIM;
    const int NI = in_sizes[1] / DDIM;
    const int E  = in_sizes[2];
    const int nu64 = NU * DDIM, ni64 = NI * DDIM;

    char* w = (char*)d_ws;
    size_t off = 0;
    auto grab = [&](size_t nfloats) -> float* {
        float* p = (float*)(w + off);
        off = alignup(off + nfloats * sizeof(float));
        return p;
    };
    float* u_a  = grab(nu64);  float* i_a  = grab(ni64);
    float* u_b  = grab(nu64);  float* i_b  = grab(ni64);
    float* usum = grab(nu64);  float* isum = grab(ni64);
    float* a_e  = grab(E);
    float* exu  = grab(E);     float* exi  = grab(E);
    float* umax = grab(NU);    float* imax = grab(NI);
    float* uden = grab(NU);    float* iden = grab(NI);
    float* acc  = grab(2);
    (void)ws_size; (void)n_in; (void)out_size;

    const int T = 256;
    const int n_emb = (nu64 > ni64 ? nu64 : ni64);
    const int gEmb  = (n_emb + T - 1) / T;
    const int gE    = (E + T - 1) / T;
    const int gE64  = (int)(((long)E * DDIM + T - 1) / T);
    const int gTile = (E + 127) / 128;             // 8 waves/block * 16 edges/wave
    const int gLoss = (NU + 127) / 128;

    k_init<<<gEmb, T, 0, stream>>>(uemb, iemb, u_a, usum, i_a, isum, acc, nu64, ni64);

    float *ucur = u_a, *icur = i_a, *unxt = u_b, *inxt = i_b;
    for (int layer = 0; layer < 2; ++layer) {
        k_reset<<<gEmb, T, 0, stream>>>(unxt, inxt, umax, uden, imax, iden,
                                        NU, NI, nu64, ni64);
        k_pass1<<<gTile, T, 0, stream>>>(ucur, icur, uidx, iidx, a_e, umax, imax, E);
        k_pass2<<<gE, T, 0, stream>>>(a_e, uidx, iidx, umax, imax, uden, iden,
                                      exu, exi, E);
        k_pass3<<<gE64, T, 0, stream>>>(exu, exi, uden, iden, uidx, iidx,
                                        ucur, icur, unxt, inxt, E);
        k_accum<<<gEmb, T, 0, stream>>>(usum, isum, unxt, inxt, nu64, ni64);
        float* t;
        t = ucur; ucur = unxt; unxt = t;
        t = icur; icur = inxt; inxt = t;
    }

    k_loss<<<gLoss, T, 0, stream>>>(usum, isum, posi, negi, acc, NU);
    k_final<<<1, 64, 0, stream>>>(acc, out, NU);
}
